// AltBlock_10866267259495
// MI455X (gfx1250) — matrix operand order repaired and checked by reading, not compiled, not run on hardware
//
#include <hip/hip_runtime.h>
#include <hip/hip_bf16.h>

// ---------------------------------------------------------------------------
// MI455X (gfx1250) fused transformer block.
//  - f16 WMMA (v_wmma_f32_16x16x32_f16) for all GEMMs, f32 accumulate.
//  - One workgroup (512 threads / 16 wave32) per (batch, 64-token block).
//  - Multi-accumulator waves: one A fragment feeds 2-8 WMMAs (ILP to cover
//    WMMA->VALU hazards, 8x fewer A-fragment LDS reads).
//  - Stage-0 x load uses GLOBAL_LOAD_ASYNC_TO_LDS_B128 (ASYNCcnt) when the
//    toolchain exposes the builtin.
// ---------------------------------------------------------------------------

typedef __attribute__((ext_vector_type(16))) _Float16 v16h;
typedef __attribute__((ext_vector_type(8)))  _Float16 v8h;
typedef __attribute__((ext_vector_type(8)))  float    v8f;
typedef __attribute__((ext_vector_type(4)))  float    v4f;
typedef __attribute__((ext_vector_type(4)))  int      v4i;

#define DIMC   512
#define LSEQ   4096
#define NHEADS 8
#define HDIM   64
#define BLK    64

// padded LDS strides (avoid bank conflicts on transposed accesses)
#define LDA  520   // f16 row stride for 512-wide buffers
#define LDR  516   // f32 row stride for 512-wide buffers
#define LDH  72    // f16 row stride for 64-wide head buffers
#define LDSS 68    // f32 row stride for 64x64 score buffer

#define SMEM_BYTES (2*(64*LDA*2) + 64*LDR*4 + 3*(64*LDH*2) + 64*LDSS*4 + 64*LDH*2)
// = 66560*2 + 132096 + 27648 + 17408 + 9216 = 319488 bytes dynamic LDS

#if defined(__gfx1250__) && __has_builtin(__builtin_amdgcn_global_load_async_to_lds_b128)
#define HAVE_ASYNC_LDS 1
typedef __attribute__((address_space(1))) v4i g_v4i;   // global int4
typedef __attribute__((address_space(3))) v4i l_v4i;   // LDS int4
#else
#define HAVE_ASYNC_LDS 0
#endif

// ---- WMMA helpers ----------------------------------------------------------

__device__ __forceinline__ v8f wmma_f16(v16h a, v16h b, v8f c) {
    return __builtin_amdgcn_wmma_f32_16x16x32_f16(
        /*neg_a=*/false, a, /*neg_b=*/false, b,
        /*c_mod=*/(short)0, c, /*reuse_a=*/false, /*reuse_b=*/false);
}

// A fragment (16x32, f16): lane holds row m; element j -> K = kk + 8*half + j (j<8)
// and K = kk + 16 + 8*half + (j-8) (j>=8).  Two contiguous 16B loads.
__device__ __forceinline__ v16h frag_a(const _Float16* base, int ldm, int row,
                                       int kk, int half) {
    const _Float16* p = base + row * ldm + kk + 8 * half;
    union { v16h v; v8h h[2]; } u;
    u.h[0] = *(const v8h*)(p);
    u.h[1] = *(const v8h*)(p + 16);
    return u.v;
}

// B fragment (32x16, f16): lane holds column n = rowptr's row; element j ->
// K = kk + 8*half + j (j<8) and K = kk + 16 + 8*half + (j-8) (j>=8).  Two 16B loads.
__device__ __forceinline__ v16h frag_b(const _Float16* rowptr, int kk, int half) {
    const _Float16* p = rowptr + kk + 8 * half;
    union { v16h v; v8h h[2]; } u;
    u.h[0] = *(const v8h*)(p);
    u.h[1] = *(const v8h*)(p + 16);
    return u.v;
}

// ---- weight prep: fp32 (K x N) row-major  ->  f16 (N x K) row-major --------

__global__ void transpose_to_f16(const float* __restrict__ in,
                                 _Float16* __restrict__ out, int K, int N) {
    int idx = blockIdx.x * blockDim.x + threadIdx.x;
    int total = K * N;
    if (idx < total) {
        int n = idx / K, k = idx - n * K;
        out[idx] = (_Float16)in[k * N + n];
    }
}

// ---- fused block kernel ----------------------------------------------------

__global__ __launch_bounds__(512, 1)
void altblock_fused(const float* __restrict__ x,
                    const _Float16* __restrict__ wqkvT,   // 1536 x 512
                    const float* __restrict__ b_qkv,
                    const _Float16* __restrict__ wprojT,  // 512 x 512
                    const float* __restrict__ b_proj,
                    const float* __restrict__ ln1_g, const float* __restrict__ ln1_b,
                    const float* __restrict__ ln2_g, const float* __restrict__ ln2_b,
                    const _Float16* __restrict__ wffn1T,  // 2048 x 512
                    const float* __restrict__ b_ffn1,
                    const _Float16* __restrict__ wffn2T,  // 512 x 1024
                    const float* __restrict__ b_ffn2,
                    const float* __restrict__ attn_scale, const float* __restrict__ attn_bias,
                    const float* __restrict__ mlp_scale,  const float* __restrict__ mlp_bias,
                    float* __restrict__ out)
{
    extern __shared__ char smem[];
    _Float16* sA = (_Float16*)smem;              // 64 x LDA  f16 : x_block / h (A operand)
    _Float16* sO = sA + 64 * LDA;                // 64 x LDA  f16 : attention output
    float*    sR = (float*)(sO + 64 * LDA);      // 64 x LDR  f32 : residual stream
    _Float16* sQ = (_Float16*)(sR + 64 * LDR);   // 64 x LDH  f16 : Q  (reused as GLU tile)
    _Float16* sK = sQ + 64 * LDH;                // 64 x LDH  f16 : K
    _Float16* sV = sK + 64 * LDH;                // 64 x LDH  f16 : V transposed [dim][tok]
    float*    sS = (float*)(sV + 64 * LDH);      // 64 x LDSS f32 : scores
    _Float16* sP = (_Float16*)(sS + 64 * LDSS);  // 64 x LDH  f16 : probs
    __shared__ float sMu[64], sRs[64];

    const int tid  = threadIdx.x;
    const int wave = tid >> 5;
    const int lane = tid & 31;
    const int m16  = lane & 15;
    const int hv   = (lane >> 4) & 1;
    const int b    = blockIdx.x >> 6;            // batch
    const int l0   = (blockIdx.x & 63) * BLK;    // token block start

    // ---- stage 0: x block (B,D,L layout) -> sA f16 [tok][dim] --------------
#if HAVE_ASYNC_LDS
    {
        // async DMA the 128KB f32 block into the (currently free) sR region,
        // laid out [dim][tok] (contiguous per-row); tracked by ASYNCcnt.
        float* sX = sR;
        const char* gb = (const char*)(x + ((size_t)b * DIMC) * LSEQ + l0);
        char*       lb = (char*)sX;
        for (int c = tid; c < DIMC * 4; c += 512) {   // 2048 x 64B segments
            int dim = c >> 2, q = c & 3;              // 4 x 64B per 256B row
            size_t go = (size_t)dim * (LSEQ * 4) + (size_t)q * 64;
            size_t lo = (size_t)dim * 256 + (size_t)q * 64;
            __builtin_amdgcn_global_load_async_to_lds_b128(
                (g_v4i*)(gb + go),      (l_v4i*)(lb + lo),      0, 0);
            __builtin_amdgcn_global_load_async_to_lds_b128(
                (g_v4i*)(gb + go + 16), (l_v4i*)(lb + lo + 16), 0, 0);
            __builtin_amdgcn_global_load_async_to_lds_b128(
                (g_v4i*)(gb + go + 32), (l_v4i*)(lb + lo + 32), 0, 0);
            __builtin_amdgcn_global_load_async_to_lds_b128(
                (g_v4i*)(gb + go + 48), (l_v4i*)(lb + lo + 48), 0, 0);
        }
        asm volatile("s_wait_asynccnt 0x0" ::: "memory");
        __syncthreads();
        for (int t = 0; t < BLK; ++t)                 // tid == dim
            sA[t * LDA + tid] = (_Float16)sX[tid * 64 + t];
    }
#else
    {
        const float* xp = x + ((size_t)(b * DIMC + tid)) * LSEQ + l0;  // tid == dim
        #pragma unroll
        for (int t = 0; t < BLK; t += 4) {
            v4f v = *(const v4f*)(xp + t);
            sA[(t + 0) * LDA + tid] = (_Float16)v.x;
            sA[(t + 1) * LDA + tid] = (_Float16)v.y;
            sA[(t + 2) * LDA + tid] = (_Float16)v.z;
            sA[(t + 3) * LDA + tid] = (_Float16)v.w;
        }
    }
#endif
    __syncthreads();

    // ---- block attention, per head -----------------------------------------
    for (int h = 0; h < NHEADS; ++h) {
        // QKV GEMM: wave (mtq, nj) computes one Q, one K, one V 16x16 tile,
        // all sharing the same A fragment (3 independent accumulators).
        {
            const int mtq = wave & 3;
            const int nj  = wave >> 2;               // 0..3
            const int c0  = h * 192 + nj * 16 + m16; // Q column in w_qkv
            const _Float16* bq = wqkvT + (size_t)c0 * DIMC;
            const _Float16* bk = bq + (size_t)64 * DIMC;
            const _Float16* bv = bq + (size_t)128 * DIMC;
            v8f aq = {}, ak = {}, av = {};
            for (int kk = 0; kk < DIMC; kk += 32) {
                v16h a = frag_a(sA, LDA, mtq * 16 + m16, kk, hv);
                aq = wmma_f16(a, frag_b(bq, kk, hv), aq);
                ak = wmma_f16(a, frag_b(bk, kk, hv), ak);
                av = wmma_f16(a, frag_b(bv, kk, hv), av);
            }
            float biq = b_qkv[c0], bik = b_qkv[c0 + 64], biv = b_qkv[c0 + 128];
            const int colq = nj * 16 + m16;
            #pragma unroll
            for (int r = 0; r < 8; ++r) {
                int tok = mtq * 16 + r + 8 * hv;
                sQ[tok * LDH + colq] = (_Float16)(aq[r] + biq);
                sK[tok * LDH + colq] = (_Float16)(ak[r] + bik);
            }
            union { v8h h8; v4f f4; } pk;            // V transposed [dim][tok]
            #pragma unroll
            for (int r = 0; r < 8; ++r) pk.h8[r] = (_Float16)(av[r] + biv);
            *(v4f*)&sV[colq * LDH + mtq * 16 + 8 * hv] = pk.f4;
        }
        __syncthreads();

        // S = Q @ K^T * d^-0.5 : 16 tiles, 1/wave
        {
            int mt = wave & 3, nt = wave >> 2;
            v8f acc = {};
            for (int kk = 0; kk < HDIM; kk += 32)
                acc = wmma_f16(frag_a(sQ, LDH, mt * 16 + m16, kk, hv),
                               frag_b(sK + (nt * 16 + m16) * LDH, kk, hv), acc);
            #pragma unroll
            for (int r = 0; r < 8; ++r)
                sS[(mt * 16 + r + 8 * hv) * LDSS + nt * 16 + m16] = acc[r] * 0.125f;
        }
        __syncthreads();

        // softmax rows (64 rows, one thread each)
        if (tid < 64) {
            float* row = sS + tid * LDSS;
            float mx = row[0];
            for (int k = 1; k < BLK; ++k) mx = fmaxf(mx, row[k]);
            float s = 0.f;
            for (int k = 0; k < BLK; ++k) { float e = __expf(row[k] - mx); row[k] = e; s += e; }
            float inv = 1.f / s;
            _Float16* prow = sP + tid * LDH;
            for (int k = 0; k < BLK; ++k) prow[k] = (_Float16)(row[k] * inv);
        }
        __syncthreads();

        // O_head = P @ V : 16 tiles, 1/wave ; write into sO columns [h*64, h*64+64)
        {
            int mt = wave & 3, nt = wave >> 2;
            v8f acc = {};
            for (int kk = 0; kk < BLK; kk += 32)
                acc = wmma_f16(frag_a(sP, LDH, mt * 16 + m16, kk, hv),
                               frag_b(sV + (nt * 16 + m16) * LDH, kk, hv), acc);
            #pragma unroll
            for (int r = 0; r < 8; ++r)
                sO[(mt * 16 + r + 8 * hv) * LDA + h * HDIM + nt * 16 + m16] = (_Float16)acc[r];
        }
        __syncthreads();
    }

    // ---- proj + attn scale/bias + residual (x re-read: L2-resident) --------
    // Wave owns M-tile (wave&3) and 8 N-tiles: one A fragment feeds 8 WMMAs.
    {
        const int mtp = wave & 3;
        const int nb8 = (wave >> 2) * 8;
        v8f acc[8];
        #pragma unroll
        for (int i = 0; i < 8; ++i) acc[i] = (v8f){};
        for (int kk = 0; kk < DIMC; kk += 32) {
            v16h a = frag_a(sO, LDA, mtp * 16 + m16, kk, hv);
            #pragma unroll
            for (int i = 0; i < 8; ++i) {
                const _Float16* brow = wprojT + (size_t)((nb8 + i) * 16 + m16) * DIMC;
                acc[i] = wmma_f16(a, frag_b(brow, kk, hv), acc[i]);
            }
        }
        #pragma unroll
        for (int i = 0; i < 8; ++i) {
            int dim = (nb8 + i) * 16 + m16;
            float bp = b_proj[dim], sc = attn_scale[dim], sb = attn_bias[dim];
            const float* xp = x + ((size_t)(b * DIMC + dim)) * LSEQ + l0 + mtp * 16 + 8 * hv;
            v4f x0 = *(const v4f*)xp;
            v4f x1 = *(const v4f*)(xp + 4);
            float xr[8] = {x0.x, x0.y, x0.z, x0.w, x1.x, x1.y, x1.z, x1.w};
            #pragma unroll
            for (int r = 0; r < 8; ++r)
                sR[(mtp * 16 + r + 8 * hv) * LDR + dim] = (acc[i][r] + bp) * sc + sb + xr[r];
        }
    }
    __syncthreads();

    // ---- LN1: stats per token, then normalize -> sR (f32) and sA (f16) -----
    if (tid < 64) {
        const float* row = sR + tid * LDR;
        float s = 0.f, ss = 0.f;
        for (int j = 0; j < DIMC; ++j) { float v = row[j]; s += v; ss += v * v; }
        float mu  = s * (1.f / DIMC);
        float var = ss * (1.f / DIMC) - mu * mu;
        sMu[tid] = mu; sRs[tid] = rsqrtf(var + 1e-5f);
    }
    __syncthreads();
    {
        float g = ln1_g[tid], bb = ln1_b[tid];   // tid == dim
        for (int t = 0; t < BLK; ++t) {
            float v = (sR[t * LDR + tid] - sMu[t]) * sRs[t] * g + bb;
            sR[t * LDR + tid] = v;               // f32 h for residual 2
            sA[t * LDA + tid] = (_Float16)v;     // f16 h for FFN GEMMs
        }
    }
    __syncthreads();

    // ---- GLU-FFN: stream 16 column tiles of 64; FFN2 accum in registers ----
    v8f macc[8];
    #pragma unroll
    for (int i = 0; i < 8; ++i) macc[i] = (v8f){};
    _Float16* sG = sQ;                           // reuse head scratch: 64 x LDH
    const int mt = wave & 3;
    const int nq = wave >> 2;                    // 0..3

    for (int jt = 0; jt < 16; ++jt) {
        // FFN1 out + gate tiles (shared A) -> SiLU-GLU -> f16 tile in sG
        {
            int co = jt * 64 + nq * 16 + m16;    // 0..1023
            const _Float16* bo = wffn1T + (size_t)co * DIMC;
            const _Float16* bg = wffn1T + (size_t)(1024 + co) * DIMC;
            v8f ao = {}, ag = {};
            for (int kk = 0; kk < DIMC; kk += 32) {
                v16h a = frag_a(sA, LDA, mt * 16 + m16, kk, hv);
                ao = wmma_f16(a, frag_b(bo, kk, hv), ao);
                ag = wmma_f16(a, frag_b(bg, kk, hv), ag);
            }
            float b1o = b_ffn1[co], b1g = b_ffn1[1024 + co];
            #pragma unroll
            for (int r = 0; r < 8; ++r) {
                float o = ao[r] + b1o, g = ag[r] + b1g;
                float glu = o * (g / (1.f + __expf(-g)));
                sG[(mt * 16 + r + 8 * hv) * LDH + nq * 16 + m16] = (_Float16)glu;
            }
        }
        __syncthreads();
        // FFN2 partial: one A fragment feeds 8 accumulators
        #pragma unroll
        for (int kk = 0; kk < 64; kk += 32) {
            v16h a = frag_a(sG, LDH, mt * 16 + m16, kk, hv);
            #pragma unroll
            for (int i = 0; i < 8; ++i) {
                const _Float16* brow =
                    wffn2T + (size_t)((nq * 8 + i) * 16 + m16) * 1024 + jt * 64;
                macc[i] = wmma_f16(a, frag_b(brow, kk, hv), macc[i]);
            }
        }
        __syncthreads();
    }

    // FFN2 epilogue: bias + mlp scale/bias + residual(h) -> sR
    #pragma unroll
    for (int i = 0; i < 8; ++i) {
        int col = (nq * 8 + i) * 16 + m16;
        float b2 = b_ffn2[col], ms = mlp_scale[col], mb = mlp_bias[col];
        #pragma unroll
        for (int r = 0; r < 8; ++r) {
            int tok = mt * 16 + r + 8 * hv;
            sR[tok * LDR + col] = (macc[i][r] + b2) * ms + mb + sR[tok * LDR + col];
        }
    }
    __syncthreads();

    // ---- LN2 stats ----
    if (tid < 64) {
        const float* row = sR + tid * LDR;
        float s = 0.f, ss = 0.f;
        for (int j = 0; j < DIMC; ++j) { float v = row[j]; s += v; ss += v * v; }
        float mu  = s * (1.f / DIMC);
        float var = ss * (1.f / DIMC) - mu * mu;
        sMu[tid] = mu; sRs[tid] = rsqrtf(var + 1e-5f);
    }
    __syncthreads();

    // ---- LN2 apply + transposed store back to (B, D, L); token-coalesced ----
    {
        int tok  = tid & 63;
        int dgrp = tid >> 6;                     // 0..7 -> 64 dims each
        float mu = sMu[tok], rs = sRs[tok];
        for (int dd = 0; dd < 64; ++dd) {
            int dim = dgrp * 64 + dd;
            float v = (sR[tok * LDR + dim] - mu) * rs * ln2_g[dim] + ln2_b[dim];
            out[((size_t)(b * DIMC + dim)) * LSEQ + l0 + tok] = v;
        }
    }
}

// ---------------------------------------------------------------------------

extern "C" void kernel_launch(void* const* d_in, const int* in_sizes, int n_in,
                              void* d_out, int out_size, void* d_ws, size_t ws_size,
                              hipStream_t stream) {
    const float* x          = (const float*)d_in[0];
    const float* w_qkv      = (const float*)d_in[1];
    const float* b_qkv      = (const float*)d_in[2];
    const float* w_proj     = (const float*)d_in[3];
    const float* b_proj     = (const float*)d_in[4];
    const float* ln1_g      = (const float*)d_in[5];
    const float* ln1_b      = (const float*)d_in[6];
    const float* ln2_g      = (const float*)d_in[7];
    const float* ln2_b      = (const float*)d_in[8];
    const float* w_ffn1     = (const float*)d_in[9];
    const float* b_ffn1     = (const float*)d_in[10];
    const float* w_ffn2     = (const float*)d_in[11];
    const float* b_ffn2     = (const float*)d_in[12];
    const float* attn_scale = (const float*)d_in[13];
    const float* attn_bias  = (const float*)d_in[14];
    const float* mlp_scale  = (const float*)d_in[15];
    const float* mlp_bias   = (const float*)d_in[16];
    float* out = (float*)d_out;

    // workspace: pre-transposed f16 weights (N x K row-major)
    _Float16* wqkvT  = (_Float16*)d_ws;                  // 1536 x 512
    _Float16* wprojT = wqkvT  + (size_t)1536 * 512;      // 512 x 512
    _Float16* wffn1T = wprojT + (size_t)512  * 512;      // 2048 x 512
    _Float16* wffn2T = wffn1T + (size_t)2048 * 512;      // 512 x 1024

    auto launch_tr = [&](const float* in, _Float16* o, int K, int N) {
        int total = K * N;
        transpose_to_f16<<<(total + 255) / 256, 256, 0, stream>>>(in, o, K, N);
    };
    launch_tr(w_qkv,  wqkvT,  512,  1536);
    launch_tr(w_proj, wprojT, 512,  512);
    launch_tr(w_ffn1, wffn1T, 512,  2048);
    launch_tr(w_ffn2, wffn2T, 1024, 512);

    // 8 batches * 64 token-blocks = 512 workgroups, 512 threads each
    altblock_fused<<<dim3(512), dim3(512), SMEM_BYTES, stream>>>(
        x, wqkvT, b_qkv, wprojT, b_proj, ln1_g, ln1_b, ln2_g, ln2_b,
        wffn1T, b_ffn1, wffn2T, b_ffn2, attn_scale, attn_bias,
        mlp_scale, mlp_bias, out);
}
